// WindowAttention_12446815224085
// MI455X (gfx1250) — matrix operand order repaired and checked by reading, not compiled, not run on hardware
//
#include <hip/hip_runtime.h>

typedef __attribute__((ext_vector_type(16))) _Float16 v16h;
typedef __attribute__((ext_vector_type(8)))  _Float16 v8h;
typedef __attribute__((ext_vector_type(4)))  _Float16 v4h;
typedef __attribute__((ext_vector_type(8)))  float    v8f;
typedef __attribute__((ext_vector_type(4)))  float    v4f;

#define SCALE 0.17677669529663687f  // HEAD_DIM^-0.5, HEAD_DIM=32

// Build a 16-half A/B fragment from two contiguous 8-half groups in LDS.
static __device__ __forceinline__ v16h ld_frag(const _Float16* p0, const _Float16* p1) {
  v8h lo = *(const v8h*)p0;
  v8h hi = *(const v8h*)p1;
  v16h r;
#pragma unroll
  for (int i = 0; i < 8; ++i) { r[i] = lo[i]; r[i + 8] = hi[i]; }
  return r;
}

// ---------------------------------------------------------------------------
// Kernel 1: windowed QKV projection GEMM.
//   M = 64 tokens (one window per block.y), N = 256 slice (block.x of 6),
//   K = 512, f16 WMMA with f32 accumulate. Epilogue adds bias, scales Q,
//   stores f16 panels [window][head][token][dim] into workspace.
// ---------------------------------------------------------------------------
__global__ __launch_bounds__(256) void qkv_gemm_kernel(
    const float* __restrict__ x, const float* __restrict__ qkv_w,
    const float* __restrict__ qkv_b,
    _Float16* __restrict__ q_ws, _Float16* __restrict__ k_ws,
    _Float16* __restrict__ v_ws)
{
  constexpr int AS = 72, BS = 72;            // LDS strides (halves), mult of 8
  __shared__ _Float16 lds_a[64 * AS];        // 64 tokens x 64 K-chunk
  __shared__ _Float16 lds_b[256 * BS];       // 256 N rows x 64 K-chunk

  const int tid  = threadIdx.x;
  const int lane = tid & 31;
  const int wave = tid >> 5;                 // 0..7, owns 32-wide N strip
  const int tileN = blockIdx.x;              // 0..5
  const int wdw   = blockIdx.y;              // 0..511
  const int nb = wdw >> 6;
  const int widx = wdw & 63;
  const int wy = widx >> 3, wx = widx & 7;
  const int n_base = tileN * 256;
  const int g16 = lane >> 4, l16 = lane & 15;

  v8f zero8 = {0.f,0.f,0.f,0.f,0.f,0.f,0.f,0.f};
  v8f acc[4][2];
#pragma unroll
  for (int mf = 0; mf < 4; ++mf)
#pragma unroll
    for (int nf = 0; nf < 2; ++nf) acc[mf][nf] = zero8;

  for (int k0 = 0; k0 < 512; k0 += 64) {
    // A tile: gather window tokens from x, f32 -> f16
#pragma unroll
    for (int e = 0; e < 4; ++e) {
      int idx = tid + e * 256;               // float4 units, 0..1023
      int row = idx >> 4;                    // token 0..63
      int c4  = idx & 15;
      int ty = row >> 3, tx = row & 7;
      int h = wy * 8 + ty, w = wx * 8 + tx;
      v4f d = *(const v4f*)(x + (((size_t)nb * 64 + h) * 64 + w) * 512 + k0 + c4 * 4);
      v4h hd; hd[0] = (_Float16)d[0]; hd[1] = (_Float16)d[1];
              hd[2] = (_Float16)d[2]; hd[3] = (_Float16)d[3];
      *(v4h*)&lds_a[row * AS + c4 * 4] = hd;
    }
    // B tile: qkv_w rows (out = part @ W^T, so W rows are B columns' K-vectors)
#pragma unroll
    for (int e = 0; e < 16; ++e) {
      int idx = tid + e * 256;               // 0..4095
      int n  = idx >> 4;
      int c4 = idx & 15;
      v4f d = *(const v4f*)(qkv_w + ((size_t)(n_base + n)) * 512 + k0 + c4 * 4);
      v4h hd; hd[0] = (_Float16)d[0]; hd[1] = (_Float16)d[1];
              hd[2] = (_Float16)d[2]; hd[3] = (_Float16)d[3];
      *(v4h*)&lds_b[n * BS + c4 * 4] = hd;
    }
    __syncthreads();

#pragma unroll
    for (int kk = 0; kk < 64; kk += 32) {
      v16h bf[2];
#pragma unroll
      for (int nf = 0; nf < 2; ++nf) {
        int nrow = wave * 32 + nf * 16 + l16;
        const _Float16* bp = &lds_b[nrow * BS + kk + g16 * 8];
        bf[nf] = ld_frag(bp, bp + 16);
      }
#pragma unroll
      for (int mf = 0; mf < 4; ++mf) {
        const _Float16* ap = &lds_a[(mf * 16 + l16) * AS + kk + g16 * 8];
        v16h af = ld_frag(ap, ap + 16);
#pragma unroll
        for (int nf = 0; nf < 2; ++nf)
          acc[mf][nf] = __builtin_amdgcn_wmma_f32_16x16x32_f16(
              false, af, false, bf[nf], (short)0, acc[mf][nf], false, false);
      }
    }
    __syncthreads();
  }

  // Epilogue: each wave's 32-wide N strip is exactly one head of Q, K or V.
  const int n_wave = n_base + wave * 32;
  const int qkvi = n_wave >> 9;              // 0=Q 1=K 2=V
  const int head = (n_wave & 511) >> 5;
  _Float16* dst = (qkvi == 0) ? q_ws : (qkvi == 1 ? k_ws : v_ws);
  const float sc = (qkvi == 0) ? SCALE : 1.0f;
#pragma unroll
  for (int nf = 0; nf < 2; ++nf) {
    int d = nf * 16 + l16;                   // head dim 0..31
    float bv = qkv_b[n_wave + d];
#pragma unroll
    for (int mf = 0; mf < 4; ++mf)
#pragma unroll
      for (int v = 0; v < 8; ++v) {
        int tok = mf * 16 + v + g16 * 8;
        float val = (acc[mf][nf][v] + bv) * sc;
        dst[(((size_t)wdw * 16 + head) * 64 + tok) * 32 + d] = (_Float16)val;
      }
  }
}

// ---------------------------------------------------------------------------
// Kernel 2: per-(window, head) attention. 1 wave / block.
//   S = Q K^T (16 WMMAs) -> LDS -> register-resident softmax -> P(f16, in
//   the S buffer) -> O = P V (16 WMMAs) -> scattered to output layout.
// ---------------------------------------------------------------------------
__global__ __launch_bounds__(32) void attn_kernel(
    const _Float16* __restrict__ q_ws, const _Float16* __restrict__ k_ws,
    const _Float16* __restrict__ v_ws, const float* __restrict__ bias_table,
    float* __restrict__ out)
{
  constexpr int QS = 40, VS = 72, PS = 72;   // LDS strides (halves), mult of 8
  __shared__ _Float16 lds_q[64 * QS];        // [token][dim]
  __shared__ _Float16 lds_k[64 * QS];        // [token][dim]
  __shared__ _Float16 lds_vt[32 * VS];       // [dim][token] (transposed V)
  __shared__ float lds_s[64 * 64];           // scores f32; reused as P (f16)
  __shared__ float lds_bias[228];            // relative-position bias column

  const int lane = threadIdx.x;
  const int head = blockIdx.x;
  const int wdw  = blockIdx.y;
  const int nb = wdw >> 6, widx = wdw & 63;
  const int wy = widx >> 3, wx = widx & 7;
  const int g16 = lane >> 4, l16 = lane & 15;

  const size_t pbase = ((size_t)wdw * 16 + head) * (64 * 32);
  const v8h* qg = (const v8h*)(q_ws + pbase);
  const v8h* kg = (const v8h*)(k_ws + pbase);
  const v8h* vg = (const v8h*)(v_ws + pbase);
#pragma unroll
  for (int e = 0; e < 8; ++e) {
    int g = lane + e * 32;                   // 8-half group 0..255
    int tok = g >> 2;
    int dim = (g & 3) * 8;
    v8h qd = qg[g]; *(v8h*)&lds_q[tok * QS + dim] = qd;
    v8h kd = kg[g]; *(v8h*)&lds_k[tok * QS + dim] = kd;
    v8h vd = vg[g];
#pragma unroll
    for (int j = 0; j < 8; ++j) lds_vt[(dim + j) * VS + tok] = vd[j];
  }
#pragma unroll
  for (int e = 0; e < 8; ++e) {
    int i = lane + e * 32;
    if (i < 225) lds_bias[i] = bias_table[i * 16 + head];
  }
  __syncthreads();

  v8f zero8 = {0.f,0.f,0.f,0.f,0.f,0.f,0.f,0.f};

  // S = Q @ K^T  (K dim = 32 = one WMMA per 16x16 tile)
  v16h aQ[4];
#pragma unroll
  for (int mf = 0; mf < 4; ++mf) {
    const _Float16* p = &lds_q[(mf * 16 + l16) * QS + g16 * 8];
    aQ[mf] = ld_frag(p, p + 16);
  }
#pragma unroll
  for (int nf = 0; nf < 4; ++nf) {
    const _Float16* p = &lds_k[(nf * 16 + l16) * QS + g16 * 8];
    v16h bK = ld_frag(p, p + 16);
#pragma unroll
    for (int mf = 0; mf < 4; ++mf) {
      v8f s = __builtin_amdgcn_wmma_f32_16x16x32_f16(
          false, aQ[mf], false, bK, (short)0, zero8, false, false);
#pragma unroll
      for (int v = 0; v < 8; ++v)
        lds_s[(mf * 16 + v + g16 * 8) * 64 + nf * 16 + l16] = s[v];
    }
  }
  __syncthreads();

  // Register-resident softmax: each lane owns rows (lane) and (lane+32).
  v4f row[2][16];
#pragma unroll
  for (int e = 0; e < 2; ++e) {
    int r = lane + e * 32;
#pragma unroll
    for (int c = 0; c < 16; ++c) row[e][c] = *(const v4f*)&lds_s[r * 64 + c * 4];
  }
  __syncthreads();                           // all S reads done; buffer reused

  _Float16* lds_p = (_Float16*)lds_s;        // P (f16), stride PS=72
#pragma unroll
  for (int e = 0; e < 2; ++e) {
    int r = lane + e * 32;
    int iy = r >> 3, ix = r & 7;
    int hy = wy * 8 + iy, hx = wx * 8 + ix;
    int ci = ((hy < 56) ? 0 : (hy < 60 ? 1 : 2)) * 3 +
             ((hx < 56) ? 0 : (hx < 60 ? 1 : 2));
    float mx = -1e30f;
#pragma unroll
    for (int c = 0; c < 16; ++c)
#pragma unroll
      for (int q = 0; q < 4; ++q) {
        int j = c * 4 + q;
        int jy = j >> 3, jx = j & 7;
        int gy = wy * 8 + jy, gx = wx * 8 + jx;
        int cj = ((gy < 56) ? 0 : (gy < 60 ? 1 : 2)) * 3 +
                 ((gx < 56) ? 0 : (gx < 60 ? 1 : 2));
        float v = row[e][c][q] + lds_bias[(iy - jy + 7) * 15 + (ix - jx + 7)] +
                  ((ci != cj) ? -100.0f : 0.0f);
        row[e][c][q] = v;
        mx = fmaxf(mx, v);
      }
    float sum = 0.0f;
#pragma unroll
    for (int c = 0; c < 16; ++c)
#pragma unroll
      for (int q = 0; q < 4; ++q) {
        float ev = __expf(row[e][c][q] - mx);
        row[e][c][q] = ev;
        sum += ev;
      }
    float inv = 1.0f / sum;
#pragma unroll
    for (int c = 0; c < 16; ++c) {
      v4h p4;
#pragma unroll
      for (int q = 0; q < 4; ++q) p4[q] = (_Float16)(row[e][c][q] * inv);
      *(v4h*)&lds_p[r * PS + c * 4] = p4;
    }
  }
  __syncthreads();

  // O = P @ V  (K dim = 64 tokens -> two WMMA steps)
  v8f o[4][2];
#pragma unroll
  for (int mf = 0; mf < 4; ++mf)
#pragma unroll
    for (int nf = 0; nf < 2; ++nf) o[mf][nf] = zero8;
#pragma unroll
  for (int kk = 0; kk < 64; kk += 32) {
    v16h bV[2];
#pragma unroll
    for (int nf = 0; nf < 2; ++nf) {
      const _Float16* p = &lds_vt[(nf * 16 + l16) * VS + kk + g16 * 8];
      bV[nf] = ld_frag(p, p + 16);
    }
#pragma unroll
    for (int mf = 0; mf < 4; ++mf) {
      const _Float16* p = &lds_p[(mf * 16 + l16) * PS + kk + g16 * 8];
      v16h aP = ld_frag(p, p + 16);
#pragma unroll
      for (int nf = 0; nf < 2; ++nf)
        o[mf][nf] = __builtin_amdgcn_wmma_f32_16x16x32_f16(
            false, aP, false, bV[nf], (short)0, o[mf][nf], false, false);
    }
  }

  // Scatter to (n, H, W, C) layout (window inverse permutation).
#pragma unroll
  for (int mf = 0; mf < 4; ++mf)
#pragma unroll
    for (int nf = 0; nf < 2; ++nf)
#pragma unroll
      for (int v = 0; v < 8; ++v) {
        int tok = mf * 16 + v + g16 * 8;
        int d = nf * 16 + l16;
        int ty = tok >> 3, tx = tok & 7;
        int h = wy * 8 + ty, w = wx * 8 + tx;
        out[(((size_t)nb * 64 + h) * 64 + w) * 512 + head * 32 + d] = o[mf][nf][v];
      }
}

// ---------------------------------------------------------------------------
extern "C" void kernel_launch(void* const* d_in, const int* in_sizes, int n_in,
                              void* d_out, int out_size, void* d_ws, size_t ws_size,
                              hipStream_t stream) {
  (void)in_sizes; (void)n_in; (void)out_size; (void)ws_size;
  const float* x          = (const float*)d_in[0];
  const float* qkv_w      = (const float*)d_in[1];
  const float* qkv_b      = (const float*)d_in[2];
  const float* bias_table = (const float*)d_in[3];
  float* out = (float*)d_out;

  // f16 Q/K/V panels: 3 * 512 windows * 16 heads * 64 tok * 32 dim = 96 MB
  _Float16* q_ws = (_Float16*)d_ws;
  _Float16* k_ws = q_ws + (size_t)512 * 16 * 64 * 32;
  _Float16* v_ws = k_ws + (size_t)512 * 16 * 64 * 32;

  dim3 g1(6, 512);    // (N tile of 256, window)
  qkv_gemm_kernel<<<g1, 256, 0, stream>>>(x, qkv_w, qkv_b, q_ws, k_ws, v_ws);

  dim3 g2(16, 512);   // (head, window)
  attn_kernel<<<g2, 32, 0, stream>>>(q_ws, k_ws, v_ws, bias_table, out);
}
